// MultiHeadSelfAttention_2705829396836
// MI455X (gfx1250) — compile-verified, matrix operand order checked by reading, not run on hardware
//
#include <hip/hip_runtime.h>
#include <hip/hip_bf16.h>

// ---------------------------------------------------------------------------
// MHA forward for MI455X (gfx1250, wave32, WMMA + TDM).
// Pipeline: cvt(f32->f16) -> QKV GEMM (WMMA) -> flash attention (WMMA, K tile
//           staged by Tensor Data Mover, row-sums via ones-WMMA) ->
//           output GEMM (WMMA, f32 out).
// Workspace requirement: 5*B*S*D + 4*D*D f16 elements = 92,274,688 bytes.
// ---------------------------------------------------------------------------

typedef __attribute__((ext_vector_type(16))) _Float16 v16h;
typedef __attribute__((ext_vector_type(8)))  _Float16 v8h;
typedef __attribute__((ext_vector_type(8)))  float    v8f;
typedef __attribute__((ext_vector_type(4)))  unsigned int u32x4;
typedef __attribute__((ext_vector_type(8)))  int       i32x8;
typedef __attribute__((ext_vector_type(4)))  int       i32x4;

constexpr int B_  = 4;
constexpr int S_  = 2048;
constexpr int D_  = 1024;
constexpr int H_  = 16;
constexpr int HD_ = 64;
constexpr int M_  = B_ * S_;          // 8192 rows of x
constexpr int NX  = M_ * D_;          // elements in x / Q / K / V / attn
constexpr int NW  = D_ * D_;          // elements per weight matrix

__device__ __forceinline__ v8f wmma16x16x32(v16h a, v16h b, v8f c) {
  return __builtin_amdgcn_wmma_f32_16x16x32_f16(
      /*neg_a=*/false, a, /*neg_b=*/false, b,
      /*c_mod=*/(short)0, c, /*reuse_a=*/false, /*reuse_b=*/false);
}

// A-fragment of a 16x32 (MxK) row-major tile with row stride ld.
// Also the B-fragment loader when the B matrix is staged TRANSPOSED
// (N x K row-major): the lane/element mapping is identical, and the
// contiguous 8-half runs lower to ds_load_b128.
__device__ __forceinline__ v16h load_a_frag(const _Float16* p0, int ld) {
  const int lane = threadIdx.x & 31;
  const int row  = lane & 15;
  const int koff = (lane >> 4) * 8;
  const _Float16* p = p0 + row * ld;
  v16h a;
#pragma unroll
  for (int i = 0; i < 8; ++i) a[i] = p[koff + i];
#pragma unroll
  for (int i = 0; i < 8; ++i) a[8 + i] = p[16 + koff + i];
  return a;
}

// ---------------------------------------------------------------------------
// fp32 -> fp16 conversion
// ---------------------------------------------------------------------------
__global__ __launch_bounds__(256) void cvt_f32_f16_kernel(
    const float* __restrict__ src, _Float16* __restrict__ dst, int n) {
  int i = blockIdx.x * blockDim.x + threadIdx.x;
  int stride = gridDim.x * blockDim.x;
  for (; i < n; i += stride) dst[i] = (_Float16)src[i];
}

// ---------------------------------------------------------------------------
// Tiled WMMA GEMM: C[M_, D_] = X[M_, D_] * W[D_, D_] + bias
// Block tile 128x64, K step 32, 256 threads = 8 waves.
// W tile staged TRANSPOSED in LDS so every fragment read is contiguous b128.
// ---------------------------------------------------------------------------
constexpr int TM  = 128, TN = 64, TK = 32;
constexpr int LDA = TK + 8;   // 40 halves = 80 B rows (16-B aligned)

template <bool HEAD_OUT>
__global__ __launch_bounds__(256) void gemm_wmma_kernel(
    const _Float16* __restrict__ X,
    const _Float16* __restrict__ W,
    const float*    __restrict__ bias,
    _Float16*       __restrict__ OutH,   // used if HEAD_OUT
    float*          __restrict__ OutF) { // used if !HEAD_OUT
  __shared__ _Float16 As [TM][LDA];
  __shared__ _Float16 BsT[TN][LDA];      // W tile stored [n][k]

  const int tid  = threadIdx.x;
  const int wave = tid >> 5;
  const int lane = tid & 31;
  const int m0 = blockIdx.x * TM;
  const int n0 = blockIdx.y * TN;

  v8f acc[4] = {};

  for (int k0 = 0; k0 < D_; k0 += TK) {
    // Stage A: 128x32 halves, 2 x b128 per thread.
#pragma unroll
    for (int c = 0; c < 2; ++c) {
      int chunk = tid * 2 + c;          // 0..511
      int r  = chunk >> 2;              // 0..127
      int c8 = (chunk & 3) * 8;         // 0,8,16,24
      *(v8h*)&As[r][c8] = *(const v8h*)(X + (size_t)(m0 + r) * D_ + k0 + c8);
    }
    // Stage B transposed: read 8 contiguous n at one k, scatter to [n][k].
    {
      int r  = tid >> 3;                // k row 0..31
      int c8 = (tid & 7) * 8;           // n offset 0..56
      v8h w = *(const v8h*)(W + (size_t)(k0 + r) * D_ + n0 + c8);
#pragma unroll
      for (int j = 0; j < 8; ++j) BsT[c8 + j][r] = w[j];
    }
    if (k0 + TK < D_) {                 // gfx1250 global_prefetch of next tiles
      __builtin_prefetch(X + (size_t)(m0 + (tid >> 1)) * D_ + k0 + TK, 0, 1);
      __builtin_prefetch(W + (size_t)(k0 + TK + (tid >> 3)) * D_ + n0, 0, 1);
    }
    __syncthreads();

    v16h a = load_a_frag(&As[wave * 16][0], LDA);
#pragma unroll
    for (int t = 0; t < 4; ++t) {
      v16h b = load_a_frag(&BsT[t * 16][0], LDA);   // B-frag from transposed tile
      acc[t] = wmma16x16x32(a, b, acc[t]);
    }
    __syncthreads();
  }

  // Epilogue. C layout: lane -> col = lane&15, rows = (lane>>4)*8 + vgpr.
  const int col = lane & 15;
  const int rb  = (lane >> 4) * 8;
#pragma unroll
  for (int t = 0; t < 4; ++t) {
    const int n  = n0 + t * 16 + col;
    const float bv = bias[n];
#pragma unroll
    for (int r = 0; r < 8; ++r) {
      const int m = m0 + wave * 16 + rb + r;
      const float val = acc[t][r] + bv;
      if (HEAD_OUT) {
        const int bb = m >> 11;         // m / S_
        const int s  = m & (S_ - 1);
        const int h  = n >> 6;          // n / HD_
        const int hd = n & (HD_ - 1);
        OutH[(((size_t)bb * H_ + h) * S_ + s) * HD_ + hd] = (_Float16)val;
      } else {
        OutF[(size_t)m * D_ + n] = val;
      }
    }
  }
}

// ---------------------------------------------------------------------------
// Flash attention: one block = one (b,h) pair x 128 query rows, 8 waves.
// K tile staged by the Tensor Data Mover into padded LDS; V tile staged
// transposed so PV fragment reads are contiguous b128. Softmax row sums are
// computed on the matrix pipe (P x ones via WMMA -> sums land in C-layout);
// only the row max needs the bpermute shuffle path. The 1/sqrt(HD) scale is
// folded into the Q fragments (exact: power of two in f16).
// ---------------------------------------------------------------------------
__global__ __launch_bounds__(256) void attn_wmma_kernel(
    const _Float16* __restrict__ Qh,   // [B,H,S,HD]
    const _Float16* __restrict__ Kh,   // [B,H,S,HD]
    const _Float16* __restrict__ Vh,   // [B,H,S,HD]
    _Float16*       __restrict__ AO) { // [M_, D_]  (b*S+s major, h*64+hd minor)
  __shared__ _Float16 Ks [64][72];     // [key][hd], pitch 72 (TDM pad)
  __shared__ _Float16 VsT[64][72];     // [hd][key]  (transposed staging)
  __shared__ _Float16 Ps [8][16][72];  // per-wave P tile (C-layout -> A-layout)

  const int tid  = threadIdx.x;
  const int wave = tid >> 5;
  const int lane = tid & 31;
  const int qt = blockIdx.x & (S_ / 128 - 1);  // query tile 0..15
  const int bh = blockIdx.x >> 4;              // b*H + h, 0..63
  const size_t hoff = (size_t)bh * S_ * HD_;
  const int q0 = qt * 128 + wave * 16;

  const int col  = lane & 15;
  const int rb   = (lane >> 4) * 8;
  const int koff = rb;

  // ---- TDM descriptor (wave-uniform; only global_addr varies per k-tile) --
  const unsigned lds_k = (unsigned)(uintptr_t)(&Ks[0][0]);   // low 32 = LDS offset
  i32x8 g1;
  g1[0] = (1 << 16) | (1 << 20) | (4 << 22) | (3 << 25);
  g1[1] = (HD_ & 0xFFFF) << 16;                 // tensor_dim0[15:0] in bits 63:48
  g1[2] = (S_ & 0xFFFF) << 16;                  // tensor_dim1[15:0] in bits 95:80
  g1[3] = (64 << 16);                           // tile_dim0=64 in bits 127:112
  g1[4] = 64;                                   // tile_dim1=64 in bits 143:128
  g1[5] = HD_;                                  // tensor_dim0_stride[31:0]
  g1[6] = 0;
  g1[7] = 0;
  const i32x4 gz4 = {0, 0, 0, 0};
  const i32x8 gz8 = {0, 0, 0, 0, 0, 0, 0, 0};

  // All-ones B fragment: P(16x64) x ones(64x16) puts row sums in every
  // column of the C tile -> per-lane sums already in C layout.
  v16h onesf;
#pragma unroll
  for (int i = 0; i < 16; ++i) onesf[i] = (_Float16)1.0f;

  // Q fragments (HD=64 -> 2 fragments of 16x32), pre-scaled by 1/sqrt(HD).
  v16h qf[2];
  {
    const _Float16* qp = Qh + hoff + (size_t)(q0 + col) * HD_;
    const _Float16 scl = (_Float16)0.125f;
#pragma unroll
    for (int h2 = 0; h2 < 2; ++h2) {
#pragma unroll
      for (int i = 0; i < 8; ++i) qf[h2][i]     = qp[h2 * 32 + koff + i] * scl;
#pragma unroll
      for (int i = 0; i < 8; ++i) qf[h2][8 + i] = qp[h2 * 32 + 16 + koff + i] * scl;
    }
  }

  v8f o[4] = {};
  float mrow[8], lrow[8];
#pragma unroll
  for (int r = 0; r < 8; ++r) { mrow[r] = -1e30f; lrow[r] = 0.0f; }

  for (int kt = 0; kt < S_; kt += 64) {
    // ---- K tile via Tensor Data Mover (wave 0 issues, then waits) ----
    if (wave == 0) {
      const unsigned long long ga =
          (unsigned long long)(uintptr_t)(Kh + hoff + (size_t)kt * HD_);
      u32x4 g0;
      g0[0] = 1u;                                   // count=1
      g0[1] = lds_k;                                // lds_addr
      g0[2] = (unsigned)(ga & 0xFFFFFFFFu);         // global_addr[31:0]
      g0[3] = (unsigned)((ga >> 32) & 0x01FFFFFFu) | (2u << 30); // [56:32]|type=2
      __builtin_amdgcn_tensor_load_to_lds(g0, g1, gz4, gz4, gz8, 0);
      __builtin_amdgcn_s_wait_tensorcnt(0);
    }
    // ---- V tile, transposed staging: [key][hd] -> VsT[hd][key] ----
#pragma unroll
    for (int c = 0; c < 2; ++c) {
      int chunk = tid * 2 + c;                 // 0..511
      int r  = chunk >> 3;                     // key 0..63
      int c8 = (chunk & 7) * 8;                // hd offset 0..56
      v8h vv = *(const v8h*)(Vh + hoff + (size_t)(kt + r) * HD_ + c8);
#pragma unroll
      for (int j = 0; j < 8; ++j) VsT[c8 + j][r] = vv[j];
    }
    __syncthreads();

    // scores = (Q/sqrt(HD)) * K^T
    v8f sc[4] = {};
#pragma unroll
    for (int t = 0; t < 4; ++t) {
#pragma unroll
      for (int h2 = 0; h2 < 2; ++h2) {
        v16h bf = load_a_frag(&Ks[t * 16][h2 * 32], 72);
        sc[t] = wmma16x16x32(qf[h2], bf, sc[t]);
      }
    }

    // online softmax: row max across the 16-lane C-layout group
    float mnew[8], corr[8];
#pragma unroll
    for (int r = 0; r < 8; ++r) {
      float v = fmaxf(fmaxf(sc[0][r], sc[1][r]), fmaxf(sc[2][r], sc[3][r]));
#pragma unroll
      for (int mk = 1; mk <= 8; mk <<= 1) v = fmaxf(v, __shfl_xor(v, mk, 32));
      mnew[r] = fmaxf(mrow[r], v);
      corr[r] = __expf(mrow[r] - mnew[r]);
      mrow[r] = mnew[r];
    }
    // p = exp(s - m), straight into the per-wave LDS P tile (f16)
#pragma unroll
    for (int t = 0; t < 4; ++t)
#pragma unroll
      for (int r = 0; r < 8; ++r)
        Ps[wave][rb + r][t * 16 + col] = (_Float16)__expf(sc[t][r] - mnew[r]);
    // rescale running output
#pragma unroll
    for (int t = 0; t < 4; ++t)
#pragma unroll
      for (int r = 0; r < 8; ++r) o[t][r] *= corr[r];
    __syncthreads();

    // P fragments (reused by both the ones-reduction and PV)
    v16h pa[2];
    pa[0] = load_a_frag(&Ps[wave][0][0],  72);
    pa[1] = load_a_frag(&Ps[wave][0][32], 72);

    // row sums on the matrix pipe: sums = P x ones, lands in C layout
    v8f st = {};
    st = wmma16x16x32(pa[0], onesf, st);
    st = wmma16x16x32(pa[1], onesf, st);
#pragma unroll
    for (int r = 0; r < 8; ++r) lrow[r] = lrow[r] * corr[r] + st[r];

    // O += P * V    (B-frag of V from transposed tile == A-frag load)
#pragma unroll
    for (int h2 = 0; h2 < 2; ++h2) {
#pragma unroll
      for (int t = 0; t < 4; ++t) {
        v16h vb = load_a_frag(&VsT[t * 16][h2 * 32], 72);
        o[t] = wmma16x16x32(pa[h2], vb, o[t]);
      }
    }
    __syncthreads();
  }

  // Finalize: O /= l, write f16 into [M_, D_] layout for the output GEMM.
  const int bb = bh >> 4;
  const int h  = bh & (H_ - 1);
#pragma unroll
  for (int t = 0; t < 4; ++t) {
#pragma unroll
    for (int r = 0; r < 8; ++r) {
      const int s = q0 + rb + r;
      const int d = h * HD_ + t * 16 + col;
      AO[((size_t)bb * S_ + s) * D_ + d] = (_Float16)(o[t][r] / lrow[r]);
    }
  }
}

// ---------------------------------------------------------------------------
// launcher
// ---------------------------------------------------------------------------
extern "C" void kernel_launch(void* const* d_in, const int* in_sizes, int n_in,
                              void* d_out, int out_size, void* d_ws, size_t ws_size,
                              hipStream_t stream) {
  (void)in_sizes; (void)n_in; (void)out_size; (void)ws_size;

  const float* x  = (const float*)d_in[0];
  const float* Wq = (const float*)d_in[1];
  const float* bq = (const float*)d_in[2];
  const float* Wk = (const float*)d_in[3];
  const float* bk = (const float*)d_in[4];
  const float* Wv = (const float*)d_in[5];
  const float* bv = (const float*)d_in[6];
  const float* Wo = (const float*)d_in[7];
  const float* bo = (const float*)d_in[8];
  float* out = (float*)d_out;

  _Float16* ws  = (_Float16*)d_ws;
  _Float16* xh  = ws;                 // [M_, D_]
  _Float16* wqh = xh  + (size_t)NX;   // [D_, D_]
  _Float16* wkh = wqh + (size_t)NW;
  _Float16* wvh = wkh + (size_t)NW;
  _Float16* woh = wvh + (size_t)NW;
  _Float16* qh  = woh + (size_t)NW;   // [B,H,S,HD]
  _Float16* kh  = qh  + (size_t)NX;
  _Float16* vh  = kh  + (size_t)NX;
  _Float16* ah  = vh  + (size_t)NX;   // [M_, D_] attention output

  // 1) fp32 -> fp16
  cvt_f32_f16_kernel<<<2048, 256, 0, stream>>>(x,  xh,  NX);
  cvt_f32_f16_kernel<<<512,  256, 0, stream>>>(Wq, wqh, NW);
  cvt_f32_f16_kernel<<<512,  256, 0, stream>>>(Wk, wkh, NW);
  cvt_f32_f16_kernel<<<512,  256, 0, stream>>>(Wv, wvh, NW);
  cvt_f32_f16_kernel<<<512,  256, 0, stream>>>(Wo, woh, NW);

  // 2) Q/K/V projections (WMMA GEMM, head-major f16 out)
  dim3 gg(M_ / TM, D_ / TN);   // (64, 16)
  gemm_wmma_kernel<true><<<gg, 256, 0, stream>>>(xh, wqh, bq, qh, nullptr);
  gemm_wmma_kernel<true><<<gg, 256, 0, stream>>>(xh, wkh, bk, kh, nullptr);
  gemm_wmma_kernel<true><<<gg, 256, 0, stream>>>(xh, wvh, bv, vh, nullptr);

  // 3) attention: B*H*(S/128) = 1024 blocks
  attn_wmma_kernel<<<B_ * H_ * (S_ / 128), 256, 0, stream>>>(qh, kh, vh, ah);

  // 4) output projection (f32 out + bias)
  gemm_wmma_kernel<false><<<gg, 256, 0, stream>>>(ah, woh, bo, nullptr, out);
}
